// LSTM_70763881169224
// MI455X (gfx1250) — matrix operand order repaired and checked by reading, not compiled, not run on hardware
//
#include <hip/hip_runtime.h>
#include <stdint.h>

#define BATCH  64
#define TSEQ   512
#define INDIM  512
#define HID    1024
#define GATES  4096     // 4*HID
#define ODIM   512
#define KTOT   1536     // INDIM + HID
#define KTILES 48       // KTOT/32
#define GRIDB  16
#define BLOCK  256
#define NWAVE  8

typedef unsigned short u16;
typedef __attribute__((ext_vector_type(16))) __bf16 v16bf;
typedef __attribute__((ext_vector_type(8)))  float  v8f;

static __device__ inline u16 f32_to_bf16(float f) {
  unsigned u = __float_as_uint(f);
  unsigned r = u + 0x7FFFu + ((u >> 16) & 1u);   // round-to-nearest-even
  return (u16)(r >> 16);
}

static __device__ inline v8f wmma_bf16(v16bf a, v16bf b, v8f c) {
  return __builtin_amdgcn_wmma_f32_16x16x32_bf16(
      false, a, false, b, (short)0, c, false, false);
}

// A-fragment (16-bit A 16x32 row-major): lane holds K chunk [k0, k0+8) and [k0+16, k0+24)
static __device__ inline v16bf load_a(const u16* p) {
  union { struct { uint4 lo; uint4 hi; } q; v16bf v; } u;
  u.q.lo = *(const uint4*)(p);
  u.q.hi = *(const uint4*)(p + 16);
  return u.v;
}

// B-fragment from packed buffer: 32 contiguous bytes per lane
static __device__ inline v16bf load_b(const u16* p) {
  union { struct { uint4 lo; uint4 hi; } q; v16bf v; } u;
  u.q.lo = *(const uint4*)(p);
  u.q.hi = *(const uint4*)(p + 8);
  return u.v;
}

// fast activations: v_exp_f32 + v_rcp_f32, branch-free, correct saturation
static __device__ inline float sigmoid_fast(float x) {
  return __builtin_amdgcn_rcpf(1.0f + __expf(-x));
}
static __device__ inline float tanh_fast(float x) {
  // 1 - 2/(exp(2x)+1):  x->+inf => 1-0 = 1;  x->-inf => 1-2 = -1
  return 1.0f - 2.0f * __builtin_amdgcn_rcpf(__expf(2.0f * x) + 1.0f);
}

// ---------------- init: zero h buffer 0 + barrier counter ----------------
__global__ void lstm_init(unsigned* hbuf_u32, unsigned* bar) {
  const int n = (BATCH * HID * 2) / 4;  // bytes of h-buffer 0 as u32
  for (int i = blockIdx.x * blockDim.x + threadIdx.x; i < n;
       i += gridDim.x * blockDim.x)
    hbuf_u32[i] = 0u;
  if (blockIdx.x == 0 && threadIdx.x == 0) *bar = 0u;
}

// ---------------- convert x (row-major f32 -> row-major bf16) ----------------
__global__ void lstm_cvt_x(const float* __restrict__ x, u16* __restrict__ xbf, int n) {
  for (int i = blockIdx.x * blockDim.x + threadIdx.x; i < n;
       i += gridDim.x * blockDim.x)
    xbf[i] = f32_to_bf16(x[i]);
}

// ---------------- pack W_i2h [KTOT x GATES] into WMMA B-fragment order ----------------
// Wp[((kt*256 + nt)*32 + lane)*16 + e] = bf16(W[k*GATES + n])
//   n = nt*16 + (lane&15),  k = kt*32 + (lane>>4)*8 + (e&7) + (e>>3)*16
__global__ void lstm_pack_wi(const float* __restrict__ W, u16* __restrict__ Wp) {
  const int total = KTILES * 256 * 32 * 16;
  for (int i = blockIdx.x * blockDim.x + threadIdx.x; i < total;
       i += gridDim.x * blockDim.x) {
    int e    = i & 15;
    int lane = (i >> 4) & 31;
    int t2   = i >> 9;
    int nt   = t2 & 255;
    int kt   = t2 >> 8;
    int n = nt * 16 + (lane & 15);
    int k = kt * 32 + (lane >> 4) * 8 + (e & 7) + (e >> 3) * 16;
    Wp[i] = f32_to_bf16(W[(size_t)k * GATES + n]);
  }
}

// ---------------- pack W_h2o [HID x ODIM], 32 ktiles x 32 ntiles ----------------
__global__ void lstm_pack_wo(const float* __restrict__ W, u16* __restrict__ Wp) {
  const int total = 32 * 32 * 32 * 16;
  for (int i = blockIdx.x * blockDim.x + threadIdx.x; i < total;
       i += gridDim.x * blockDim.x) {
    int e    = i & 15;
    int lane = (i >> 4) & 31;
    int t2   = i >> 9;
    int nt   = t2 & 31;
    int kt   = t2 >> 5;
    int n = nt * 16 + (lane & 15);
    int k = kt * 32 + (lane >> 4) * 8 + (e & 7) + (e >> 3) * 16;
    Wp[i] = f32_to_bf16(W[(size_t)k * ODIM + n]);
  }
}

// ---------------- persistent LSTM recurrence ----------------
__global__ __launch_bounds__(BLOCK) void lstm_persistent(
    const u16* __restrict__ xbf, const u16* __restrict__ wip,
    const u16* __restrict__ wop, const float* __restrict__ b_i2h,
    const float* __restrict__ b_h2o, u16* __restrict__ hbuf,
    unsigned* bar, float* __restrict__ out) {
  const int lane  = threadIdx.x & 31;
  const int wv    = threadIdx.x >> 5;
  const int gw    = blockIdx.x * NWAVE + wv;   // [0,128)
  const int mgrp  = gw & 1;                    // two 16-row M tiles each
  const int nslot = gw >> 1;                   // [0,64) hidden 16-col slice
  const int m0    = mgrp * 32;
  const int lcol  = lane & 15;
  const int khalf = lane >> 4;
  const int j0    = nslot * 16;
  const int col   = j0 + lcol;

  const float bias_i = b_i2h[0 * HID + col];
  const float bias_f = b_i2h[1 * HID + col];
  const float bias_g = b_i2h[2 * HID + col];
  const float bias_o = b_i2h[3 * HID + col];

  const int mA0 = m0 + lcol;
  const int mA1 = m0 + 16 + lcol;

  // packed-B base pointer per gate section (kt = 0)
  const u16* wb[4];
#pragma unroll
  for (int g = 0; g < 4; ++g) {
    int ng = g * 64 + nslot;                      // global 16-col ntile in [0,256)
    wb[g] = wip + ((size_t)ng * 32 + lane) * 16;
  }
  const size_t wstride = (size_t)256 * 32 * 16;   // elems per ktile

  v8f z = {};
  v8f creg[2];
  creg[0] = z;
  creg[1] = z;

#pragma unroll 1
  for (int t = 0; t < TSEQ; ++t) {
    const u16* hcur = hbuf + (size_t)(t & 1) * (BATCH * HID);
    u16*       hnxt = hbuf + (size_t)((t + 1) & 1) * (BATCH * HID);

    v8f acc[2][4];
#pragma unroll
    for (int mi = 0; mi < 2; ++mi)
#pragma unroll
      for (int g = 0; g < 4; ++g) acc[mi][g] = z;

    // ---- x contribution: K = 0..511 ----
    const u16* xr0 = xbf + ((size_t)mA0 * TSEQ + t) * INDIM + khalf * 8;
    const u16* xr1 = xbf + ((size_t)mA1 * TSEQ + t) * INDIM + khalf * 8;
#pragma unroll 4
    for (int kt = 0; kt < 16; ++kt) {
      v16bf a0 = load_a(xr0 + kt * 32);
      v16bf a1 = load_a(xr1 + kt * 32);
#pragma unroll
      for (int g = 0; g < 4; ++g) {
        v16bf bm = load_b(wb[g] + (size_t)kt * wstride);
        acc[0][g] = wmma_bf16(a0, bm, acc[0][g]);
        acc[1][g] = wmma_bf16(a1, bm, acc[1][g]);
      }
    }
    // ---- h contribution: K = 512..1535 ----
    const u16* hr0 = hcur + (size_t)mA0 * HID + khalf * 8;
    const u16* hr1 = hcur + (size_t)mA1 * HID + khalf * 8;
#pragma unroll 4
    for (int kt = 16; kt < KTILES; ++kt) {
      v16bf a0 = load_a(hr0 + (kt - 16) * 32);
      v16bf a1 = load_a(hr1 + (kt - 16) * 32);
#pragma unroll
      for (int g = 0; g < 4; ++g) {
        v16bf bm = load_b(wb[g] + (size_t)kt * wstride);
        acc[0][g] = wmma_bf16(a0, bm, acc[0][g]);
        acc[1][g] = wmma_bf16(a1, bm, acc[1][g]);
      }
    }

    // ---- elementwise LSTM update; c lives in registers; publish h as bf16 ----
#pragma unroll
    for (int mi = 0; mi < 2; ++mi) {
#pragma unroll
      for (int r = 0; r < 8; ++r) {
        float iv = sigmoid_fast(acc[mi][0][r] + bias_i);
        float fv = sigmoid_fast(acc[mi][1][r] + bias_f);
        float gv = tanh_fast(acc[mi][2][r] + bias_g);
        float ov = sigmoid_fast(acc[mi][3][r] + bias_o);
        float cn = fv * creg[mi][r] + iv * gv;
        creg[mi][r] = cn;
        float hn = ov * tanh_fast(cn);
        int mrow = m0 + mi * 16 + r + khalf * 8;   // C-layout row mapping
        hnxt[(size_t)mrow * HID + col] = f32_to_bf16(hn);
      }
    }

    // ---- device-wide barrier: flush h, count arrivals, invalidate, proceed ----
    __threadfence();       // my h stores reach device scope
    __syncthreads();       // whole block flushed
    if (threadIdx.x == 0) {
      atomicAdd(bar, 1u);
      const unsigned target = (unsigned)(t + 1) * (unsigned)GRIDB;
      while (*(volatile unsigned*)bar < target) __builtin_amdgcn_s_sleep(1);
    }
    __syncthreads();
    __threadfence();       // acquire: drop stale h lines before next read
  }

  // ---- final projection: out = h_last @ W_h2o + b_h2o  (h_last = buffer 0) ----
  {
    const int mt = gw & 3;        // M tile [0,4)
    const int nt = gw >> 2;       // N tile [0,32)
    v8f acc2 = z;
    const int mA = mt * 16 + lcol;
    const u16* hr = hbuf + (size_t)mA * HID + khalf * 8;
    const u16* wob = wop + ((size_t)nt * 32 + lane) * 16;
    const size_t wstride2 = (size_t)32 * 32 * 16;
#pragma unroll 4
    for (int kt = 0; kt < 32; ++kt) {
      v16bf a  = load_a(hr + kt * 32);
      v16bf bm = load_b(wob + (size_t)kt * wstride2);
      acc2 = wmma_bf16(a, bm, acc2);
    }
    const float bias = b_h2o[nt * 16 + lcol];
#pragma unroll
    for (int r = 0; r < 8; ++r) {
      int mrow = mt * 16 + r + khalf * 8;
      out[(size_t)mrow * ODIM + nt * 16 + lcol] = acc2[r] + bias;
    }
  }
}

extern "C" void kernel_launch(void* const* d_in, const int* in_sizes, int n_in,
                              void* d_out, int out_size, void* d_ws, size_t ws_size,
                              hipStream_t stream) {
  (void)in_sizes; (void)n_in; (void)out_size; (void)ws_size;
  const float* x     = (const float*)d_in[0];
  const float* W_i2h = (const float*)d_in[1];
  const float* b_i2h = (const float*)d_in[2];
  const float* W_h2o = (const float*)d_in[3];
  const float* b_h2o = (const float*)d_in[4];
  float* out = (float*)d_out;

  char* ws = (char*)d_ws;
  const size_t XBF = (size_t)BATCH * TSEQ * INDIM * 2;  // 33,554,432
  const size_t WIP = (size_t)KTOT * GATES * 2;          // 12,582,912
  const size_t WOP = (size_t)HID * ODIM * 2;            //  1,048,576
  const size_t HB  = (size_t)2 * BATCH * HID * 2;       //    262,144
  u16* xbf  = (u16*)(ws);
  u16* wip  = (u16*)(ws + XBF);
  u16* wop  = (u16*)(ws + XBF + WIP);
  u16* hbuf = (u16*)(ws + XBF + WIP + WOP);
  unsigned* bar = (unsigned*)(ws + XBF + WIP + WOP + HB);

  lstm_init<<<64, 256, 0, stream>>>((unsigned*)hbuf, bar);
  lstm_cvt_x<<<4096, 256, 0, stream>>>(x, xbf, BATCH * TSEQ * INDIM);
  lstm_pack_wi<<<6144, 256, 0, stream>>>(W_i2h, wip);
  lstm_pack_wo<<<512, 256, 0, stream>>>(W_h2o, wop);
  lstm_persistent<<<GRIDB, BLOCK, 0, stream>>>(xbf, wip, wop, b_i2h, b_h2o,
                                               hbuf, bar, out);
}
